// DesignerNetwork_48198122995973
// MI455X (gfx1250) — matrix operand order repaired and checked by reading, not compiled, not run on hardware
//
#include <hip/hip_runtime.h>
#include <math.h>

// ---------------------------------------------------------------------------
// MI455X (gfx1250) implementation.
//
// Strategy (latency-bound sequential GRU chains, ~90 GFLOP, ~6MB HBM traffic):
//  * grid = 64 workgroups, one per 16-row batch tile (WMMA M=16). Fully
//    independent -> no cross-WG sync.
//  * block = 64 threads = 2 wave32: wave0 runs the forward chain, wave1 the
//    backward chain (independent until the heads).
//  * All weights + rho state live in LDS (~281KB of the 320KB WGP LDS),
//    pre-converted to bf16 in a B-fragment-friendly padded layout.
//  * GRU matmuls use v_wmma_f32_16x16x32_bf16 (2 k-steps for K=60), f32
//    accumulate; gates padded to 64 cols so 16-wide N tiles never straddle
//    gate boundaries. 48 WMMA per GRU step.
//  * adj[j,i]==0 => step is exactly identity -> uniform-branch skip (~6x
//    shorter critical chain at 30% edge density).
//  * Tiny heads (omega/value/psi + softmax) in VALU from LDS.
// ---------------------------------------------------------------------------

typedef unsigned short u16;
typedef __bf16 bf16_t;
typedef bf16_t v16bf __attribute__((ext_vector_type(16)));
typedef float  v8f   __attribute__((ext_vector_type(8)));

union V16 { v16bf v; uint4 q[2]; u16 u[16]; };

// ---- LDS layout (bytes) ----------------------------------------------------
constexpr int OFF_WIHF = 0;                    // [192][64] bf16 = 24576
constexpr int OFF_WHHF = OFF_WIHF + 24576;
constexpr int OFF_WIHB = OFF_WHHF + 24576;
constexpr int OFF_WHHB = OFF_WIHB + 24576;
constexpr int OFF_WFT  = OFF_WHHB + 24576;     // [64][96] bf16 = 12288
constexpr int OFF_WBT  = OFF_WFT  + 12288;
constexpr int OFF_RHOF = OFF_WBT  + 12288;     // [32][16][64] bf16 = 65536
constexpr int OFF_RHOB = OFF_RHOF + 65536;
constexpr int OFF_BIHF = OFF_RHOB + 65536;     // [192] f32 = 768
constexpr int OFF_BHHF = OFF_BIHF + 768;
constexpr int OFF_BIHB = OFF_BHHF + 768;
constexpr int OFF_BHHB = OFF_BIHB + 768;
constexpr int OFF_BF   = OFF_BHHB + 768;       // [64] f32 = 256
constexpr int OFF_BB   = OFF_BF   + 256;
constexpr int OFF_HAS  = OFF_BB   + 256;       // [16][32] f32 = 2048
constexpr int OFF_ADJ  = OFF_HAS  + 2048;      // [32][32] f32 = 4096
constexpr int OFF_HST  = OFF_ADJ  + 4096;      // 2 x [16][64] bf16 = 4096
constexpr int OFF_XC   = OFF_HST  + 4096;      // 2 x [16][96] bf16 = 6144
constexpr int OFF_GLOB = OFF_XC   + 6144;      // [16][120] f32 = 7680
constexpr int OFF_WA   = OFF_GLOB + 7680;      // [5][120] f32 = 2400
constexpr int OFF_BA   = OFF_WA   + 2400;      // 32 (padded)
constexpr int OFF_WC   = OFF_BA   + 32;        // [120] f32 = 480
constexpr int OFF_BC   = OFF_WC   + 480;       // 16 (padded)
constexpr int OFF_WU   = OFF_BC   + 16;        // [7][120] f32 = 3360
constexpr int OFF_BU   = OFF_WU   + 3360;      // 32 (padded)
constexpr int LDS_BYTES = OFF_BU + 32;         // 287,920 B  (< 320KB WGP LDS)

// ---- small helpers ---------------------------------------------------------
__device__ __forceinline__ u16 f2bf(float f) {        // f32 -> bf16 bits (RNE)
  unsigned u = __builtin_bit_cast(unsigned, f);
  u += 0x7FFFu + ((u >> 16) & 1u);
  return (u16)(u >> 16);
}
__device__ __forceinline__ float bf2f(u16 h) {
  return __builtin_bit_cast(float, ((unsigned)h) << 16);
}
__device__ __forceinline__ float sigm(float x) { return 1.f / (1.f + __expf(-x)); }
__device__ __forceinline__ v8f splat8(float x) {
  v8f v;
#pragma unroll
  for (int i = 0; i < 8; ++i) v[i] = x;
  return v;
}
__device__ __forceinline__ v8f wmma_bf(v16bf a, v16bf b, v8f c) {
  // D = A(16x32 bf16) * B(32x16 bf16) + C(16x16 f32)
  return __builtin_amdgcn_wmma_f32_16x16x32_bf16(false, a, false, b, (short)0, c,
                                                 false, false);
}

// A fragment (16x32 bf16) from an LDS tile [16 rows][stride bf16].
// Layout: lanes 0-15 row M=lane, K = kb..kb+15;
//         lanes16-31 row M=lane-16, K = kb+16..kb+31.
__device__ __forceinline__ v16bf load_a16(const u16* tile, int stride, int kb, int lane) {
  const int row = lane & 15;
  const int kh  = (lane & 16) ? 16 : 0;
  const u16* p = tile + row * stride + kb + kh;
  V16 r;
  r.q[0] = *(const uint4*)(p);
  r.q[1] = *(const uint4*)(p + 8);
  return r.v;
}

// B fragment (32x16 bf16) from LDS weight matrix stored as [n rows][K cols]
// row-major (B[k][n] = W[n][k]).  Lane holds column n = n0 + lane%16,
// K run = kb + (lane<16 ? 0..15 : 16..31): 16 contiguous bf16 -> 2x b128.
__device__ __forceinline__ v16bf load_b16(const u16* w, int kstride, int n0, int kb, int lane) {
  const int n = n0 + (lane & 15);
  const int k = kb + ((lane & 16) ? 16 : 0);
  const u16* p = w + n * kstride + k;
  V16 r;
  r.q[0] = *(const uint4*)(p);
  r.q[1] = *(const uint4*)(p + 8);
  return r.v;
}

// One masked GRU step on a 16-batch tile:
//   h <- h + m * (GRU(x, h) - h),  m[row] = m_scale * has_col[row]
// x, h are read from LDS tiles [16][64] bf16; h kept in f32 C-fragments hc[4]
// (cols 0..63, gates padded) and mirrored back to the bf16 staging tile.
__device__ __forceinline__ void gru_step(
    const u16* xt, u16* hst, v8f hc[4],
    const u16* Wih, const u16* Whh,
    const float* bih, const float* bhh,
    float m_scale, const float* has_col /* stride 32 per row */, int lane)
{
  const int col = lane & 15;
  const int rb  = (lane & 16) ? 8 : 0;
  v16bf x0 = load_a16(xt, 64, 0, lane);
  v16bf x1 = load_a16(xt, 64, 32, lane);
  v16bf h0 = load_a16(hst, 64, 0, lane);
  v16bf h1 = load_a16(hst, 64, 32, lane);
#pragma unroll
  for (int t = 0; t < 4; ++t) {
    const int c = t * 16 + col;
    // gate-padded weight rows: r -> [0,64), u -> [64,128), n -> [128,192)
    v8f cr = splat8(bih[c] + bhh[c]);
    v8f cu = splat8(bih[64 + c] + bhh[64 + c]);
    v8f ci = splat8(bih[128 + c]);
    v8f ch = splat8(bhh[128 + c]);
    cr = wmma_bf(x0, load_b16(Wih, 64, t * 16, 0, lane), cr);
    cr = wmma_bf(x1, load_b16(Wih, 64, t * 16, 32, lane), cr);
    cr = wmma_bf(h0, load_b16(Whh, 64, t * 16, 0, lane), cr);
    cr = wmma_bf(h1, load_b16(Whh, 64, t * 16, 32, lane), cr);
    cu = wmma_bf(x0, load_b16(Wih, 64, 64 + t * 16, 0, lane), cu);
    cu = wmma_bf(x1, load_b16(Wih, 64, 64 + t * 16, 32, lane), cu);
    cu = wmma_bf(h0, load_b16(Whh, 64, 64 + t * 16, 0, lane), cu);
    cu = wmma_bf(h1, load_b16(Whh, 64, 64 + t * 16, 32, lane), cu);
    ci = wmma_bf(x0, load_b16(Wih, 64, 128 + t * 16, 0, lane), ci);
    ci = wmma_bf(x1, load_b16(Wih, 64, 128 + t * 16, 32, lane), ci);
    ch = wmma_bf(h0, load_b16(Whh, 64, 128 + t * 16, 0, lane), ch);
    ch = wmma_bf(h1, load_b16(Whh, 64, 128 + t * 16, 32, lane), ch);
    v8f hv = hc[t];
#pragma unroll
    for (int p = 0; p < 8; ++p) {
      float r = sigm(cr[p]);
      float u = sigm(cu[p]);
      float n = tanhf(ci[p] + r * ch[p]);
      float m = m_scale * has_col[(rb + p) * 32];
      float h = hv[p];
      float hn = h + m * ((1.f - u) * n + u * h - h);
      hv[p] = hn;
      hst[(rb + p) * 64 + c] = f2bf(hn);   // C-layout scatter back to staging
    }
    hc[t] = hv;
  }
}

// ---------------------------------------------------------------------------
__global__ void __launch_bounds__(64)
designer_gnn_kernel(const float* __restrict__ has, const float* __restrict__ z,
                    const float* __restrict__ dz, const int* __restrict__ adj,
                    const float* Wih_f, const float* Whh_f, const float* bih_f,
                    const float* bhh_f, const float* Wih_b, const float* Whh_b,
                    const float* bih_b, const float* bhh_b,
                    const float* Wf, const float* bff, const float* Wb,
                    const float* bbb, const float* Wa, const float* ba,
                    const float* Wc, const float* bc, const float* Wu,
                    const float* bu, float* __restrict__ out)
{
  extern __shared__ __align__(16) char smem[];
  u16*   sWihF = (u16*)(smem + OFF_WIHF);
  u16*   sWhhF = (u16*)(smem + OFF_WHHF);
  u16*   sWihB = (u16*)(smem + OFF_WIHB);
  u16*   sWhhB = (u16*)(smem + OFF_WHHB);
  u16*   sWfT  = (u16*)(smem + OFF_WFT);
  u16*   sWbT  = (u16*)(smem + OFF_WBT);
  u16*   sRhoF = (u16*)(smem + OFF_RHOF);
  u16*   sRhoB = (u16*)(smem + OFF_RHOB);
  float* sBihF = (float*)(smem + OFF_BIHF);
  float* sBhhF = (float*)(smem + OFF_BHHF);
  float* sBihB = (float*)(smem + OFF_BIHB);
  float* sBhhB = (float*)(smem + OFF_BHHB);
  float* sBf   = (float*)(smem + OFF_BF);
  float* sBb   = (float*)(smem + OFF_BB);
  float* sHas  = (float*)(smem + OFF_HAS);
  float* sAdj  = (float*)(smem + OFF_ADJ);
  u16*   sHst  = (u16*)(smem + OFF_HST);
  u16*   sXc   = (u16*)(smem + OFF_XC);
  float* sGlob = (float*)(smem + OFF_GLOB);
  float* sWa   = (float*)(smem + OFF_WA);
  float* sBa   = (float*)(smem + OFF_BA);
  float* sWc   = (float*)(smem + OFF_WC);
  float* sBc   = (float*)(smem + OFF_BC);
  float* sWu   = (float*)(smem + OFF_WU);
  float* sBu   = (float*)(smem + OFF_BU);

  const int tid  = threadIdx.x;
  const int lane = tid & 31;
  const int wave = tid >> 5;
  const int b0   = blockIdx.x * 16;

  // ---- cooperative preload: weights -> bf16 LDS (gate-padded) --------------
  {
    const float* gsrc[4] = {Wih_f, Whh_f, Wih_b, Whh_b};
    u16* gdst[4] = {sWihF, sWhhF, sWihB, sWhhB};
#pragma unroll
    for (int w = 0; w < 4; ++w) {
      const float* W = gsrc[w];
      u16* D = gdst[w];
      for (int idx = tid; idx < 192 * 64; idx += 64) {
        int rp = idx >> 6, c = idx & 63;        // padded row, K col
        int g = rp >> 6, r = rp & 63;           // gate, row-in-gate
        float v = (r < 60 && c < 60) ? W[(g * 60 + r) * 60 + c] : 0.f;
        D[idx] = f2bf(v);
      }
    }
    for (int idx = tid; idx < 64 * 96; idx += 64) {
      int n = idx / 96, k = idx % 96;           // output feat, concat index
      sWfT[idx] = f2bf((n < 60 && k < 72) ? Wf[n * 72 + k] : 0.f);
      sWbT[idx] = f2bf((n < 60 && k < 72) ? Wb[n * 72 + k] : 0.f);
    }
    for (int idx = tid; idx < 192; idx += 64) {
      int g = idx >> 6, r = idx & 63;
      bool v = (r < 60);
      sBihF[idx] = v ? bih_f[g * 60 + r] : 0.f;
      sBhhF[idx] = v ? bhh_f[g * 60 + r] : 0.f;
      sBihB[idx] = v ? bih_b[g * 60 + r] : 0.f;
      sBhhB[idx] = v ? bhh_b[g * 60 + r] : 0.f;
    }
    for (int idx = tid; idx < 64; idx += 64) {
      sBf[idx] = (idx < 60) ? bff[idx] : 0.f;
      sBb[idx] = (idx < 60) ? bbb[idx] : 0.f;
    }
    for (int idx = tid; idx < 512; idx += 64)
      sHas[idx] = has[(b0 + (idx >> 5)) * 32 + (idx & 31)];
    for (int idx = tid; idx < 1024; idx += 64)
      sAdj[idx] = adj[idx] ? 1.f : 0.f;
    for (int idx = tid; idx < 600; idx += 64) sWa[idx] = Wa[idx];
    for (int idx = tid; idx < 840; idx += 64) sWu[idx] = Wu[idx];
    for (int idx = tid; idx < 120; idx += 64) sWc[idx] = Wc[idx];
    if (tid < 5) sBa[tid] = ba[tid];
    if (tid < 7) sBu[tid] = bu[tid];
    if (tid == 0) sBc[0] = bc[0];
  }
  __syncthreads();

  // ---- per-wave direction setup -------------------------------------------
  const u16*   Wih  = wave ? sWihB : sWihF;
  const u16*   Whh  = wave ? sWhhB : sWhhF;
  const float* bih  = wave ? sBihB : sBihF;
  const float* bhh  = wave ? sBhhB : sBhhF;
  const u16*   Wcat = wave ? sWbT : sWfT;
  const float* bcat = wave ? sBb : sBf;
  u16* rho = wave ? sRhoB : sRhoF;
  u16* hst = sHst + wave * 1024;
  u16* xc  = sXc + wave * 1536;

  const int col = lane & 15;
  const int rb  = (lane & 16) ? 8 : 0;

  // ---- rho chains: fwd (wave0, i ascending) / bwd (wave1, i descending) ----
  for (int step = 0; step < 32; ++step) {
    const int i = wave ? (31 - step) : step;
    for (int idx = lane; idx < 1024; idx += 32) hst[idx] = 0;   // h = 0
    v8f hc[4];
#pragma unroll
    for (int t = 0; t < 4; ++t) hc[t] = splat8(0.f);

    for (int jj = 0; jj < 32; ++jj) {
      const int j = wave ? (31 - jj) : jj;
      const float a = wave ? sAdj[i * 32 + j] : sAdj[j * 32 + i];
      if (a != 0.f)   // adj==0 => step is exactly identity; uniform branch
        gru_step(rho + j * 1024, hst, hc, Wih, Whh, bih, bhh, a, sHas + j, lane);
    }

    // rho_i = act( [h | z_i | dz_i] @ Wcat^T + bcat ), act = tanh (fwd only)
    for (int idx = lane; idx < 16 * 96; idx += 32) {
      int row = idx / 96, c = idx % 96;
      u16 b;
      if (c < 60)      b = hst[row * 64 + c];
      else if (c < 66) b = f2bf(z[((b0 + row) * 32 + i) * 6 + (c - 60)]);
      else if (c < 72) b = f2bf(dz[((b0 + row) * 32 + i) * 6 + (c - 66)]);
      else             b = 0;
      xc[idx] = b;
    }
    v16bf a0 = load_a16(xc, 96, 0, lane);
    v16bf a1 = load_a16(xc, 96, 32, lane);
    v16bf a2 = load_a16(xc, 96, 64, lane);
#pragma unroll
    for (int t = 0; t < 4; ++t) {
      v8f cacc = splat8(bcat[t * 16 + col]);
      cacc = wmma_bf(a0, load_b16(Wcat, 96, t * 16, 0, lane), cacc);
      cacc = wmma_bf(a1, load_b16(Wcat, 96, t * 16, 32, lane), cacc);
      cacc = wmma_bf(a2, load_b16(Wcat, 96, t * 16, 64, lane), cacc);
#pragma unroll
      for (int p = 0; p < 8; ++p) {
        float v = cacc[p];
        if (!wave) v = tanhf(v);
        rho[i * 1024 + (rb + p) * 64 + t * 16 + col] = f2bf(v);
      }
    }
  }

  // ---- alpha chains --------------------------------------------------------
  {
    for (int idx = lane; idx < 1024; idx += 32) hst[idx] = 0;
    v8f hc[4];
#pragma unroll
    for (int t = 0; t < 4; ++t) hc[t] = splat8(0.f);
    if (!wave) {  // alpha_f over nodes 27..31
      for (int i = 27; i < 32; ++i)
        gru_step(rho + i * 1024, hst, hc, Wih, Whh, bih, bhh, 1.f, sHas + i, lane);
    } else {      // alpha_b over nodes 5..0
      for (int i = 5; i >= 0; --i)
        gru_step(rho + i * 1024, hst, hc, Wih, Whh, bih, bhh, 1.f, sHas + i, lane);
    }
    const int off = wave ? 60 : 0;  // glob = [alpha_f | alpha_b]
#pragma unroll
    for (int t = 0; t < 4; ++t)
#pragma unroll
      for (int p = 0; p < 8; ++p) {
        int c = t * 16 + col;
        if (c < 60) sGlob[(rb + p) * 120 + off + c] = hc[t][p];
      }
  }
  __syncthreads();

  // ---- heads: omega softmax + value (lanes 0..15 of wave0) -----------------
  if (tid < 16) {
    const int row = tid;
    float o[5], mx = -1e30f;
#pragma unroll
    for (int a = 0; a < 5; ++a) {
      float s = sBa[a];
      for (int k = 0; k < 120; ++k) s += sGlob[row * 120 + k] * sWa[a * 120 + k];
      o[a] = s;
      mx = fmaxf(mx, s);
    }
    float den = 0.f;
#pragma unroll
    for (int a = 0; a < 5; ++a) { o[a] = __expf(o[a] - mx); den += o[a]; }
    const float inv = 1.f / den;
#pragma unroll
    for (int a = 0; a < 5; ++a) out[(b0 + row) * 5 + a] = o[a] * inv;
    float v = sBc[0];
    for (int k = 0; k < 120; ++k) v += sGlob[row * 120 + k] * sWc[k];
    out[234496 + b0 + row] = v;   // value at offset 5*B + 7*32*B
  }

  // ---- heads: psi = softmax_roles(mask([rho_f|rho_b] @ Wu^T + bu)) ---------
  for (int task = tid; task < 512; task += 64) {
    const int node = task >> 4, row = task & 15;
    float acc[7];
#pragma unroll
    for (int r = 0; r < 7; ++r) acc[r] = sBu[r];
    for (int k = 0; k < 60; ++k) {
      float ff = bf2f(sRhoF[node * 1024 + row * 64 + k]);
      float fb = bf2f(sRhoB[node * 1024 + row * 64 + k]);
#pragma unroll
      for (int r = 0; r < 7; ++r)
        acc[r] += ff * sWu[r * 120 + k] + fb * sWu[r * 120 + 60 + k];
    }
    const float hv = sHas[row * 32 + node];
    float mx = -1e30f;
#pragma unroll
    for (int r = 0; r < 7; ++r) {
      acc[r] = -60.f * (1.f - hv) + acc[r] * hv;
      mx = fmaxf(mx, acc[r]);
    }
    float den = 0.f;
#pragma unroll
    for (int r = 0; r < 7; ++r) { acc[r] = __expf(acc[r] - mx); den += acc[r]; }
    const float inv = 1.f / den;
#pragma unroll
    for (int r = 0; r < 7; ++r)
      out[5120 + (b0 + row) * 224 + r * 32 + node] = acc[r] * inv;
  }
}

// ---------------------------------------------------------------------------
extern "C" void kernel_launch(void* const* d_in, const int* in_sizes, int n_in,
                              void* d_out, int out_size, void* d_ws, size_t ws_size,
                              hipStream_t stream) {
  (void)in_sizes; (void)n_in; (void)out_size; (void)d_ws; (void)ws_size;
  const float* has   = (const float*)d_in[0];
  const float* z     = (const float*)d_in[1];
  const float* dz    = (const float*)d_in[2];
  const int*   adj   = (const int*)d_in[3];     // bool adjacency as int
  const float* Wih_f = (const float*)d_in[4];
  const float* Whh_f = (const float*)d_in[5];
  const float* bih_f = (const float*)d_in[6];
  const float* bhh_f = (const float*)d_in[7];
  const float* Wih_b = (const float*)d_in[8];
  const float* Whh_b = (const float*)d_in[9];
  const float* bih_b = (const float*)d_in[10];
  const float* bhh_b = (const float*)d_in[11];
  const float* Wf    = (const float*)d_in[12];
  const float* bf    = (const float*)d_in[13];
  const float* Wb    = (const float*)d_in[14];
  const float* bb    = (const float*)d_in[15];
  const float* Wa    = (const float*)d_in[16];
  const float* ba    = (const float*)d_in[17];
  const float* Wc    = (const float*)d_in[18];
  const float* bc    = (const float*)d_in[19];
  const float* Wu    = (const float*)d_in[20];
  const float* bu    = (const float*)d_in[21];

  designer_gnn_kernel<<<dim3(64), dim3(64), LDS_BYTES, stream>>>(
      has, z, dz, adj, Wih_f, Whh_f, bih_f, bhh_f, Wih_b, Whh_b, bih_b, bhh_b,
      Wf, bf, Wb, bb, Wa, ba, Wc, bc, Wu, bu, (float*)d_out);
}
